// SelfAttention_34720515621236
// MI455X (gfx1250) — matrix operand order repaired and checked by reading, not compiled, not run on hardware
//
#include <hip/hip_runtime.h>

typedef __attribute__((ext_vector_type(16))) _Float16 v16h;
typedef __attribute__((ext_vector_type(8)))  _Float16 v8h;
typedef __attribute__((ext_vector_type(8)))  float    v8f;
typedef __attribute__((ext_vector_type(4)))  float    v4f;

#define EMBED 1024
#define HDIM  128
#define BATCH 4
#define SEQ   4096
#define ROWS  (BATCH * SEQ)   // 16384
#define PLDS_STRIDE 40        // halfs per row: 80B, 16B-aligned, breaks bank conflicts
#define KWAVES 4              // split-K waves per query tile

static __device__ __forceinline__ v16h pack16(v8h lo, v8h hi) {
    return __builtin_shufflevector(lo, hi, 0,1,2,3,4,5,6,7,8,9,10,11,12,13,14,15);
}

// ---------------------------------------------------------------------------
// Kernel 1: QKV projection, fp32 -> f16, via V_WMMA_F32_16X16X32_F16.
//   Q,K: row-major (ROWS, 128) f16.   V: transposed (B, 128, T) f16 so the
//   attention PV B-operand is a contiguous 32B load per lane (and the store
//   here vectorizes to global_store_b128, confirmed in round-1 disasm).
// grid = (ROWS/16, 6), block = 128 (4 waves); wave w handles ntile = by*4+w.
// ---------------------------------------------------------------------------
__global__ __launch_bounds__(128)
void qkv_proj_kernel(const float* __restrict__ x,
                     const float* __restrict__ Wq,
                     const float* __restrict__ Wk,
                     const float* __restrict__ Wv,
                     _Float16* __restrict__ Qh,
                     _Float16* __restrict__ Kh,
                     _Float16* __restrict__ Vt)
{
    const int lane = threadIdx.x & 31;
    const int wave = threadIdx.x >> 5;
    const int h    = lane >> 4;     // half-wave select
    const int ln   = lane & 15;

    const int mtile = blockIdx.x;                // 0..1023
    const int ntile = blockIdx.y * 4 + wave;     // 0..23
    const int mat   = ntile >> 3;                // 0:Q 1:K 2:V
    const int colb  = (ntile & 7) * 16;

    const float* W = (mat == 0) ? Wq : (mat == 1) ? Wk : Wv;

    // A-operand: row M = ln; 16-bit A 16x32 layout:
    //   as loaded here, elem i -> K = 16h + i (contiguous, same order as B)
    const float* xrow = x + (size_t)(mtile * 16 + ln) * EMBED;
    // B-operand: column n = ln; elem i -> K = 16h + i (contiguous)
    const float* wrow = W + (size_t)(colb + ln) * EMBED + 16 * h;

    v8f acc = {};
    for (int k0 = 0; k0 < EMBED; k0 += 32) {
        v4f f0 = *(const v4f*)(xrow + k0 + 16 * h);
        v4f f1 = *(const v4f*)(xrow + k0 + 16 * h + 4);
        v4f f2 = *(const v4f*)(xrow + k0 + 16 * h + 8);
        v4f f3 = *(const v4f*)(xrow + k0 + 16 * h + 12);
        v16h a;
        #pragma unroll
        for (int i = 0; i < 4; ++i) {
            a[i]      = (_Float16)f0[i];
            a[4 + i]  = (_Float16)f1[i];
            a[8 + i]  = (_Float16)f2[i];
            a[12 + i] = (_Float16)f3[i];
        }
        v4f g0 = *(const v4f*)(wrow + k0);
        v4f g1 = *(const v4f*)(wrow + k0 + 4);
        v4f g2 = *(const v4f*)(wrow + k0 + 8);
        v4f g3 = *(const v4f*)(wrow + k0 + 12);
        v16h bm;
        #pragma unroll
        for (int i = 0; i < 4; ++i) {
            bm[i]      = (_Float16)g0[i];
            bm[4 + i]  = (_Float16)g1[i];
            bm[8 + i]  = (_Float16)g2[i];
            bm[12 + i] = (_Float16)g3[i];
        }
        acc = __builtin_amdgcn_wmma_f32_16x16x32_f16(false, a, false, bm,
                                                     (short)0, acc, false, false);
    }

    // C/D layout: VGPR r, lanes 0-15 -> M=r, lanes 16-31 -> M=8+r; N = ln.
    if (mat < 2) {
        _Float16* dst = (mat == 0) ? Qh : Kh;
        #pragma unroll
        for (int r = 0; r < 8; ++r) {
            int rowg = mtile * 16 + r + 8 * h;
            dst[(size_t)rowg * HDIM + colb + ln] = (_Float16)acc[r];
        }
    } else {
        #pragma unroll
        for (int r = 0; r < 8; ++r) {
            int rowg = mtile * 16 + r + 8 * h;
            int bb = rowg >> 12;            // / SEQ
            int t  = rowg & (SEQ - 1);
            Vt[((size_t)bb * HDIM + colb + ln) * SEQ + t] = (_Float16)acc[r];
        }
    }
}

// ---------------------------------------------------------------------------
// Kernel 2: causal flash attention, split-K across 4 waves per query tile.
// grid = ROWS/16 blocks x 128 threads. Wave w handles key blocks
// s0 = w*32, w*32+128, ... (interleaved => balanced causal lengths).
// Each wave keeps partial (m, l, O) and the block combines through LDS after
// a single __syncthreads(). 4096 waves total => ~4 waves/SIMD so one wave's
// softmax VALU / K,V load latency overlaps another wave's v_wmma issue.
// ---------------------------------------------------------------------------
__global__ __launch_bounds__(128)
void attn_kernel(const _Float16* __restrict__ Qh,
                 const _Float16* __restrict__ Kh,
                 const _Float16* __restrict__ Vt,
                 float* __restrict__ out)
{
    __shared__ float     sm[KWAVES][16];
    __shared__ float     sl[KWAVES][16];
    __shared__ float     so[KWAVES][16][HDIM];             // 32 KB
    __shared__ _Float16  plds[KWAVES][16 * PLDS_STRIDE];   // 5 KB

    const int lane = threadIdx.x & 31;
    const int w    = threadIdx.x >> 5;     // split-K wave id 0..3
    const int h    = lane >> 4;
    const int ln   = lane & 15;
    const int g    = blockIdx.x;           // global query tile 0..1023
    const int b    = g >> 8;               // g / (SEQ/16)
    const int qb   = (g & 255) * 16;       // within-batch query base

    // Q tile in A layout: 4 chunks of K=32 over D=128, resident in VGPRs.
    const _Float16* qp = Qh + (size_t)(g * 16 + ln) * HDIM;
    v16h a_q[4];
    #pragma unroll
    for (int kc = 0; kc < 4; ++kc) {
        v8h lo = *(const v8h*)(qp + kc * 32 + 16 * h);
        v8h hi = *(const v8h*)(qp + kc * 32 + 16 * h + 8);
        a_q[kc] = pack16(lo, hi);
    }

    v8f zero = {};
    v8f o[8];
    #pragma unroll
    for (int dt = 0; dt < 8; ++dt) o[dt] = zero;
    float m_r[8], l_r[8];
    #pragma unroll
    for (int r = 0; r < 8; ++r) { m_r[r] = -1e30f; l_r[r] = 0.0f; }

    const float scale = 0.08838834764831845f;   // 1/sqrt(128)
    const _Float16* Kb = Kh + (size_t)b * SEQ * HDIM;
    const _Float16* Vb = Vt + (size_t)b * HDIM * SEQ;
    const int s_end = qb + 16;                  // keys < s_end participate

    for (int s0 = w * 32; s0 < s_end; s0 += KWAVES * 32) {
        // prefetch next key block of this wave (global_prefetch_b8, no counters)
        int sp = s0 + KWAVES * 32;
        if (sp < s_end) {
            int pr0 = sp + ln;        if (pr0 > SEQ - 1) pr0 = SEQ - 1;
            int pr1 = sp + 16 + ln;   if (pr1 > SEQ - 1) pr1 = SEQ - 1;
            __builtin_prefetch(Kb + (size_t)pr0 * HDIM + 16 * h, 0, 1);
            __builtin_prefetch(Kb + (size_t)pr1 * HDIM + 16 * h, 0, 1);
            int tsp = sp + 16 * h;    if (tsp > SEQ - 16) tsp = SEQ - 16;
            #pragma unroll
            for (int dt = 0; dt < 8; ++dt)
                __builtin_prefetch(Vb + (size_t)(dt * 16 + ln) * SEQ + tsp, 0, 1);
        }

        // ---- scores: two 16x16 tiles (keys s0..s0+15, s0+16..s0+31) ----
        v8f st[2];
        #pragma unroll
        for (int j = 0; j < 2; ++j) {
            int krow = s0 + 16 * j + ln;
            if (krow > SEQ - 1) krow = SEQ - 1;       // masked anyway (p==0)
            const _Float16* kp = Kb + (size_t)krow * HDIM + 16 * h;
            v8f c = zero;
            #pragma unroll
            for (int kc = 0; kc < 4; ++kc) {
                v16h bk = *(const v16h*)(kp + kc * 32);
                c = __builtin_amdgcn_wmma_f32_16x16x32_f16(false, a_q[kc], false, bk,
                                                           (short)0, c, false, false);
            }
            st[j] = c;
        }

        // ---- online softmax (row reductions within each 16-lane half) ----
        #pragma unroll
        for (int r = 0; r < 8; ++r) {
            int qrow = qb + r + 8 * h;
            float v0 = st[0][r] * scale;
            float v1 = st[1][r] * scale;
            if (s0 + ln      > qrow) v0 = -1e30f;     // causal mask
            if (s0 + 16 + ln > qrow) v1 = -1e30f;
            float mx = fmaxf(v0, v1);
            mx = fmaxf(mx, __shfl_xor(mx, 1, 32));
            mx = fmaxf(mx, __shfl_xor(mx, 2, 32));
            mx = fmaxf(mx, __shfl_xor(mx, 4, 32));
            mx = fmaxf(mx, __shfl_xor(mx, 8, 32));
            float mnew  = fmaxf(m_r[r], mx);
            float alpha = __expf(m_r[r] - mnew);
            m_r[r] = mnew;
            float p0 = __expf(v0 - mnew);
            float p1 = __expf(v1 - mnew);
            float ps = p0 + p1;
            ps += __shfl_xor(ps, 1, 32);
            ps += __shfl_xor(ps, 2, 32);
            ps += __shfl_xor(ps, 4, 32);
            ps += __shfl_xor(ps, 8, 32);
            l_r[r] = l_r[r] * alpha + ps;
            #pragma unroll
            for (int dt = 0; dt < 8; ++dt) o[dt][r] *= alpha;
            // stage P (C layout -> LDS row-major 16x32)
            plds[w][(r + 8 * h) * PLDS_STRIDE + ln]      = (_Float16)p0;
            plds[w][(r + 8 * h) * PLDS_STRIDE + 16 + ln] = (_Float16)p1;
        }

        // ---- reread P in A layout (same-wave DS ops are in-order) ----
        v8h p0v = *(const v8h*)(&plds[w][ln * PLDS_STRIDE + 16 * h]);
        v8h p1v = *(const v8h*)(&plds[w][ln * PLDS_STRIDE + 16 * h + 8]);
        v16h a_p = pack16(p0v, p1v);

        // ---- O += P @ V : 8 d-tiles, B-operand contiguous from Vt ----
        int ts = s0 + 16 * h;
        if (ts > SEQ - 16) ts = SEQ - 16;             // masked columns, p==0
        #pragma unroll
        for (int dt = 0; dt < 8; ++dt) {
            v16h bv = *(const v16h*)(Vb + (size_t)(dt * 16 + ln) * SEQ + ts);
            o[dt] = __builtin_amdgcn_wmma_f32_16x16x32_f16(false, a_p, false, bv,
                                                           (short)0, o[dt], false, false);
        }
    }

    // ---- publish per-wave partials ----
    if (ln == 0) {
        #pragma unroll
        for (int r = 0; r < 8; ++r) {
            sm[w][r + 8 * h] = m_r[r];
            sl[w][r + 8 * h] = l_r[r];
        }
    }
    #pragma unroll
    for (int r = 0; r < 8; ++r) {
        #pragma unroll
        for (int dt = 0; dt < 8; ++dt)
            so[w][r + 8 * h][dt * 16 + ln] = o[dt][r];
    }
    __syncthreads();

    // ---- combine: 128 threads, 8 threads/row, 16 cols each (coalesced) ----
    {
        const int t  = threadIdx.x;
        const int rr = t >> 3;              // 0..15
        const int cs = (t & 7) * 16;        // column segment base
        float m0 = sm[0][rr], m1 = sm[1][rr], m2 = sm[2][rr], m3 = sm[3][rr];
        float gm = fmaxf(fmaxf(m0, m1), fmaxf(m2, m3));
        float f0 = __expf(m0 - gm), f1 = __expf(m1 - gm);
        float f2 = __expf(m2 - gm), f3 = __expf(m3 - gm);
        float gl = f0 * sl[0][rr] + f1 * sl[1][rr] + f2 * sl[2][rr] + f3 * sl[3][rr];
        float inv = 1.0f / gl;
        const int rowg = g * 16 + rr;
        #pragma unroll
        for (int c = 0; c < 16; ++c) {
            float v = f0 * so[0][rr][cs + c] + f1 * so[1][rr][cs + c]
                    + f2 * so[2][rr][cs + c] + f3 * so[3][rr][cs + c];
            out[(size_t)rowg * HDIM + cs + c] = v * inv;
        }
    }
}

// ---------------------------------------------------------------------------
extern "C" void kernel_launch(void* const* d_in, const int* in_sizes, int n_in,
                              void* d_out, int out_size, void* d_ws, size_t ws_size,
                              hipStream_t stream)
{
    const float* x  = (const float*)d_in[0];
    const float* Wq = (const float*)d_in[1];
    const float* Wk = (const float*)d_in[2];
    const float* Wv = (const float*)d_in[3];
    float* out = (float*)d_out;

    // workspace: Qh (4MB) | Kh (4MB) | Vt (4MB), all f16
    _Float16* Qh = (_Float16*)d_ws;
    _Float16* Kh = Qh + (size_t)ROWS * HDIM;
    _Float16* Vt = Kh + (size_t)ROWS * HDIM;

    dim3 gproj(ROWS / 16, 6, 1);                       // 1024 x 6, 4 waves/block
    qkv_proj_kernel<<<gproj, 128, 0, stream>>>(x, Wq, Wk, Wv, Qh, Kh, Vt);

    attn_kernel<<<ROWS / 16, 128, 0, stream>>>(Qh, Kh, Vt, out);
}
